// KalmanVAE_16801912061979
// MI455X (gfx1250) — matrix operand order repaired and checked by reading, not compiled, not run on hardware
//
#include <hip/hip_runtime.h>
#include <hip/hip_bf16.h>
#include <math.h>

// Problem constants (match reference)
#define BB   256
#define SS   512
#define DDIM 64
#define LDIM 64
#define HH   256
#define NROWS (BB * SS)   // 131072

typedef __attribute__((ext_vector_type(16))) __bf16 v16bf;
typedef __attribute__((ext_vector_type(8)))  float  v8f;
typedef __attribute__((ext_vector_type(2)))  float  v2f;
typedef unsigned u32x4 __attribute__((ext_vector_type(4)));
typedef int      i32x4 __attribute__((ext_vector_type(4)));
typedef int      i32x8 __attribute__((ext_vector_type(8)));

__device__ __forceinline__ int laneid() { return (int)(threadIdx.x & 31); }

// ---- WMMA wrappers --------------------------------------------------------
__device__ __forceinline__ v8f wmma_bf16(v16bf a, v16bf b, v8f c) {
  return __builtin_amdgcn_wmma_f32_16x16x32_bf16(false, a, false, b, (short)0, c,
                                                 false, false);
}
__device__ __forceinline__ v8f wmma_f32(v2f a, v2f b, v8f c) {
  return __builtin_amdgcn_wmma_f32_16x16x4_f32(false, a, false, b, (short)0, c,
                                               false, false);
}

// ---- Fragment helpers (CDNA5 ISA 7.12.2 layouts, wave32) ------------------
// A-matrix 16x32 bf16 from LDS: lane<16 -> K={0..7,16..23}, lane>=16 -> K={8..15,24..31}
__device__ __forceinline__ v16bf frag_a_bf16(const __bf16* src, int stride) {
  const int l = laneid();
  const int m = l & 15;
  const int kb = (l < 16) ? 0 : 8;
  v16bf a;
#pragma unroll
  for (int i = 0; i < 16; ++i) {
    const int k = ((i >> 3) * 16) + kb + (i & 7);
    a[i] = src[m * stride + k];
  }
  return a;
}
// B fragment from pre-packed fragment-major bf16 (one 32B contiguous load/lane)
__device__ __forceinline__ v16bf frag_b_pk(const __bf16* base) {
  return *(const v16bf*)(base + (size_t)laneid() * 16);
}
// C/D 16x16 f32 store: VGPR r -> M=r (lanes 0-15) / r+8 (lanes 16-31), N=lane&15
__device__ __forceinline__ void store_tile_f32(float* dst, int stride, v8f d) {
  const int l = laneid();
  const int n = l & 15;
  const int mo = (l < 16) ? 0 : 8;
#pragma unroll
  for (int r = 0; r < 8; ++r) dst[(size_t)(mo + r) * stride + n] = d[r];
}

// ---- Tensor Data Mover: DMA one 64x64 f32 tile (rows of a [NROWS,64] tensor)
// from global into LDS. Call from ONE wave; follows with s_wait_tensorcnt.
__device__ __forceinline__ void tdm_load_64x64_f32(const float* gsrc, float* ldst) {
#if __has_builtin(__builtin_amdgcn_tensor_load_to_lds)
  const unsigned long long ga = (unsigned long long)gsrc;
  // Generic LDS pointer = {shared aperture, offset}; addr[31:0] is the LDS address.
  const unsigned lds = (unsigned)(unsigned long long)ldst;
  u32x4 g0;
  g0.x = 1u;                                                  // count=1 (valid D#)
  g0.y = lds;                                                 // lds_addr
  g0.z = (unsigned)ga;                                        // global_addr[31:0]
  g0.w = ((unsigned)(ga >> 32) & 0x01FFFFFFu) | 0x80000000u;  // addr[56:32] | type=2
  i32x8 g1;
  g1[0] = 0x20000;                  // wg_mask=0, data_size=2 (4 bytes)
  g1[1] = (int)(64u << 16);         // tensor_dim0 = 64   (bits 79:48, low part)
  g1[2] = 0;                        // dim0 hi, tensor_dim1 low16 (131072 -> 0)
  g1[3] = (int)((64u << 16) | 2u);  // tile_dim0=64 | tensor_dim1 hi16 = 2
  g1[4] = 64;                       // tile_dim1=64, tile_dim2=0 (2D tile)
  g1[5] = 64;                       // tensor_dim0_stride = 64
  g1[6] = 0;
  g1[7] = 0;
  i32x4 gz = {0, 0, 0, 0};          // groups 2/3 unused (tile_dim2=0)
#if __clang_major__ >= 23
  i32x8 gz8 = {0, 0, 0, 0, 0, 0, 0, 0};
  __builtin_amdgcn_tensor_load_to_lds(g0, g1, gz, gz, gz8, 0);
#else
  __builtin_amdgcn_tensor_load_to_lds(g0, g1, gz, gz, 0);
#endif
  __builtin_amdgcn_s_wait_tensorcnt(0);
#else
  for (int i = laneid(); i < 64 * 64; i += 32) ldst[i] = gsrc[i];
#endif
}

// ===========================================================================
// Kernel 1: precompute Kalman matrices (1 block, f32 VALU).
//   AQ = A@Q; innov = Hm@AQ + R; inv = innov^-1 (Gauss-Jordan);
//   K = AQ@inv; KT = K^T; G = I - Hm^T K^T; M = A^T G
// ===========================================================================
__global__ __launch_bounds__(256) void kvae_prep(
    const float* __restrict__ A, const float* __restrict__ Q,
    const float* __restrict__ Hm, const float* __restrict__ R,
    float* __restrict__ Mout, float* __restrict__ Gout,
    float* __restrict__ KTout) {
  __shared__ float sAQ[64 * 64];
  __shared__ float aug[64 * 128];
  __shared__ float fac[64];
  __shared__ float pinv;
  const int tid = threadIdx.x;

  for (int i = tid; i < 4096; i += 256) {
    const int r = i >> 6, c = i & 63;
    float s = 0.f;
    for (int k = 0; k < 64; ++k) s += A[r * 64 + k] * Q[k * 64 + c];
    sAQ[i] = s;
  }
  __syncthreads();
  for (int i = tid; i < 4096; i += 256) {
    const int r = i >> 6, c = i & 63;
    float s = R[i];
    for (int k = 0; k < 64; ++k) s += Hm[r * 64 + k] * sAQ[k * 64 + c];
    aug[r * 128 + c] = s;
    aug[r * 128 + 64 + c] = (r == c) ? 1.f : 0.f;
  }
  __syncthreads();
  for (int p = 0; p < 64; ++p) {
    if (tid == 0) pinv = 1.f / aug[p * 128 + p];
    __syncthreads();
    if (tid < 128) aug[p * 128 + tid] *= pinv;
    __syncthreads();
    if (tid < 64) fac[tid] = aug[tid * 128 + p];
    __syncthreads();
    for (int i = tid; i < 8192; i += 256) {
      const int r = i >> 7, c = i & 127;
      if (r != p) aug[i] -= fac[r] * aug[p * 128 + c];
    }
    __syncthreads();
  }
  for (int i = tid; i < 4096; i += 256) {  // K = AQ@inv (into dead left half)
    const int r = i >> 6, c = i & 63;
    float s = 0.f;
    for (int k = 0; k < 64; ++k) s += sAQ[r * 64 + k] * aug[k * 128 + 64 + c];
    aug[r * 128 + c] = s;
    KTout[c * 64 + r] = s;
  }
  __syncthreads();
  for (int i = tid; i < 4096; i += 256) {  // G = I - Hm^T K^T (reuse sAQ)
    const int r = i >> 6, c = i & 63;
    float s = (r == c) ? 1.f : 0.f;
    for (int k = 0; k < 64; ++k) s -= aug[c * 128 + k] * Hm[k * 64 + r];
    sAQ[i] = s;
    Gout[i] = s;
  }
  __syncthreads();
  for (int i = tid; i < 4096; i += 256) {  // M = A^T G
    const int r = i >> 6, c = i & 63;
    float s = 0.f;
    for (int k = 0; k < 64; ++k) s += A[k * 64 + r] * sAQ[k * 64 + c];
    Mout[i] = s;
  }
}

// ===========================================================================
// Kernel 1b: repack an f32 [K,N] weight into fragment-major bf16:
// dst[((nt*KC + kc)*32 + lane)*16 + i] = src[(kc*32 + (i>>3)*16 + kb + (i&7))*strideN + nt*16 + n]
// so each hot-loop B fragment is one contiguous 32B load per lane.
// ===========================================================================
__global__ void kvae_pack_b(const float* __restrict__ src, __bf16* __restrict__ dst,
                            int NT, int KC, int strideN) {
  const int total = NT * KC * 512;
  for (int o = blockIdx.x * blockDim.x + threadIdx.x; o < total;
       o += gridDim.x * blockDim.x) {
    const int nt = o / (KC * 512);
    const int rem = o % (KC * 512);
    const int kc = rem >> 9;
    const int r2 = rem & 511;
    const int lane = r2 >> 4;
    const int i = r2 & 15;
    const int n = lane & 15;
    const int kb = (lane < 16) ? 0 : 8;
    const int k = kc * 32 + ((i >> 3) * 16) + kb + (i & 7);
    dst[o] = (__bf16)src[(size_t)k * strideN + nt * 16 + n];
  }
}

// ===========================================================================
// Kernel 2: fused encoder + reparameterize + U precompute (64-row tile/block).
// ===========================================================================
__global__ __launch_bounds__(256) void kvae_encoder(
    const float* __restrict__ x, const float* __restrict__ eps,
    const float* __restrict__ b1, const float* __restrict__ b2,
    const __bf16* __restrict__ pw1, const __bf16* __restrict__ pw2,
    const __bf16* __restrict__ pG, const __bf16* __restrict__ pKT,
    float* __restrict__ outMean, float* __restrict__ outLogv,
    float* __restrict__ U) {
  __shared__ __bf16 sX[64 * 64];      // 8 KB  bf16 X tile
  __shared__ float  sHML[64 * 128];   // 32 KB union {f32 X stage | bf16 H | f32 ML}
  __shared__ __bf16 sZ[64 * 64];      // 8 KB  sampled z tile
  __bf16* sH  = (__bf16*)sHML;
  float*  sXf = sHML;                 // f32 staging area (first 16 KB)

  const int tid = threadIdx.x;
  const int w = tid >> 5;
  const int row0 = blockIdx.x * 64;
  const int l = laneid();
  const int n = l & 15;
  const int mo = (l < 16) ? 0 : 8;

  // ---- TDM: DMA X tile (f32) into LDS, then convert to bf16 ----
  if (tid < 32) tdm_load_64x64_f32(x + (size_t)row0 * DDIM, sXf);
  __syncthreads();
  for (int i = tid; i < 64 * 64; i += 256) sX[i] = (__bf16)sXf[i];
  __syncthreads();

  // ---- GEMM1: X[64,64] @ W1[64,256], relu -> sH (bf16) ----
#pragma unroll
  for (int nn = 0; nn < 2; ++nn) {
    const int nt = w + nn * 8;
    const v16bf bf0 = frag_b_pk(pw1 + (size_t)(nt * 2 + 0) * 512);
    const v16bf bf1 = frag_b_pk(pw1 + (size_t)(nt * 2 + 1) * 512);
    const float bias = b1[nt * 16 + n];
#pragma unroll
    for (int mt = 0; mt < 4; ++mt) {
      v8f acc = {};
      acc = wmma_bf16(frag_a_bf16(sX + mt * 16 * 64 + 0, 64), bf0, acc);
      acc = wmma_bf16(frag_a_bf16(sX + mt * 16 * 64 + 32, 64), bf1, acc);
#pragma unroll
      for (int r = 0; r < 8; ++r) {
        float v = acc[r] + bias;
        v = v > 0.f ? v : 0.f;
        sH[(mt * 16 + mo + r) * HH + nt * 16 + n] = (__bf16)v;
      }
    }
  }
  __syncthreads();

  // ---- GEMM2: H[64,256] @ W2[256,128]; B fragments hoisted over mt ----
  v8f mlacc[4] = {};
  {
    const int nt = w;  // one 16-col tile of the 128 output cols per wave
#pragma unroll
    for (int kc = 0; kc < 8; ++kc) {
      const v16bf b = frag_b_pk(pw2 + (size_t)(nt * 8 + kc) * 512);
#pragma unroll
      for (int mt = 0; mt < 4; ++mt)
        mlacc[mt] = wmma_bf16(frag_a_bf16(sH + mt * 16 * HH + kc * 32, HH), b,
                              mlacc[mt]);
    }
    const float bias = b2[nt * 16 + n];
#pragma unroll
    for (int mt = 0; mt < 4; ++mt)
#pragma unroll
      for (int r = 0; r < 8; ++r) mlacc[mt][r] += bias;
  }
  __syncthreads();  // all waves done reading sH -> reuse as f32 ML
#pragma unroll
  for (int mt = 0; mt < 4; ++mt)
    store_tile_f32(sHML + mt * 16 * 128 + w * 16, 128, mlacc[mt]);
  __syncthreads();

  // ---- write mean/logvar, reparameterize z -> sZ (bf16) ----
  for (int i = tid; i < 64 * 128; i += 256) {
    const int r = i >> 7, c = i & 127;
    const float v = sHML[i];
    const size_t rg = (size_t)(row0 + r);
    if (c < 64) outMean[rg * 64 + c] = v;
    else        outLogv[rg * 64 + (c - 64)] = v;
  }
  for (int i = tid; i < 64 * 64; i += 256) {
    const int r = i >> 6, c = i & 63;
    const float mu = sHML[r * 128 + c];
    const float lv = sHML[r * 128 + 64 + c];
    const float zz = mu + eps[(size_t)(row0 + r) * 64 + c] * expf(0.5f * lv);
    sZ[i] = (__bf16)zz;
  }
  __syncthreads();

  // ---- U = z @ G + x @ KT : two m-tiles per wave share the same n-tile ----
  {
    const int mt0 = w >> 2, mt1 = (w >> 2) + 2;
    const int nt = w & 3;
    v8f acc0 = {}, acc1 = {};
#pragma unroll
    for (int kc = 0; kc < 2; ++kc) {
      const v16bf bG = frag_b_pk(pG + (size_t)(nt * 2 + kc) * 512);
      const v16bf bK = frag_b_pk(pKT + (size_t)(nt * 2 + kc) * 512);
      acc0 = wmma_bf16(frag_a_bf16(sZ + mt0 * 16 * 64 + kc * 32, 64), bG, acc0);
      acc0 = wmma_bf16(frag_a_bf16(sX + mt0 * 16 * 64 + kc * 32, 64), bK, acc0);
      acc1 = wmma_bf16(frag_a_bf16(sZ + mt1 * 16 * 64 + kc * 32, 64), bG, acc1);
      acc1 = wmma_bf16(frag_a_bf16(sX + mt1 * 16 * 64 + kc * 32, 64), bK, acc1);
    }
    store_tile_f32(U + (size_t)(row0 + mt0 * 16) * 64 + nt * 16, 64, acc0);
    store_tile_f32(U + (size_t)(row0 + mt1 * 16) * 64 + nt * 16, 64, acc1);
  }
}

// ===========================================================================
// Kernel 3: sequential Kalman scan in full f32 (V_WMMA_F32_16X16X4_F32).
//   z[t] = z[t-1] @ M + U[t]   (in place: U becomes z_upd)
// 16 blocks x 4 waves; block owns 16 batch rows; wave owns one 16-col n-tile.
// M lives in VGPRs all 512 steps; state bounces through padded LDS.
// Prefetch U rows 4 steps ahead (global_prefetch_b8).
// ===========================================================================
__global__ __launch_bounds__(128) void kvae_scan(const float* __restrict__ Mmat,
                                                 float* __restrict__ U) {
  __shared__ float sst[16 * 68];
  const int tid = threadIdx.x;
  const int nt = tid >> 5;
  const int l = tid & 31;
  const int b0 = blockIdx.x * 16;
  const int n = l & 15;
  const int m = l & 15;
  const int mo = (l < 16) ? 0 : 8;
  const int ko = (l < 16) ? 0 : 2;

  v2f bm[16];
#pragma unroll
  for (int kc = 0; kc < 16; ++kc) {
    v2f b;
    b[0] = Mmat[(kc * 4 + ko + 0) * 64 + nt * 16 + n];
    b[1] = Mmat[(kc * 4 + ko + 1) * 64 + nt * 16 + n];
    bm[kc] = b;
  }

  for (int i = tid; i < 16 * 68; i += 128) sst[i] = 0.f;  // z0 = 0
  __syncthreads();

  for (int t = 0; t < SS; ++t) {
    if (t + 4 < SS) {
#pragma unroll
      for (int r = 0; r < 8; ++r)
        __builtin_prefetch(
            &U[((size_t)(b0 + mo + r) * SS + (t + 4)) * 64 + nt * 16 + n], 0, 1);
    }
    v2f af[16];
#pragma unroll
    for (int kc = 0; kc < 16; ++kc) {
      v2f a;
      a[0] = sst[m * 68 + kc * 4 + ko + 0];
      a[1] = sst[m * 68 + kc * 4 + ko + 1];
      af[kc] = a;
    }
    v8f c;
#pragma unroll
    for (int r = 0; r < 8; ++r)
      c[r] = U[((size_t)(b0 + mo + r) * SS + t) * 64 + nt * 16 + n];
#pragma unroll
    for (int kc = 0; kc < 16; ++kc) c = wmma_f32(af[kc], bm[kc], c);

    __syncthreads();  // all waves consumed old state before overwrite
#pragma unroll
    for (int r = 0; r < 8; ++r) {
      U[((size_t)(b0 + mo + r) * SS + t) * 64 + nt * 16 + n] = c[r];
      sst[(mo + r) * 68 + nt * 16 + n] = c[r];
    }
    __syncthreads();
  }
}

// ===========================================================================
// Kernel 4: fused decoder: x_rec = sigmoid(relu(z_upd @ W1 + b1) @ W2 + b2)
// ===========================================================================
__global__ __launch_bounds__(256) void kvae_decoder(
    const float* __restrict__ zup,
    const float* __restrict__ b1, const float* __restrict__ b2,
    const __bf16* __restrict__ pw1, const __bf16* __restrict__ pw2,
    float* __restrict__ xrec) {
  __shared__ __bf16 sZ[64 * 64];    // 8 KB
  __shared__ __bf16 sH[64 * 256];   // 32 KB (first 16 KB doubles as f32 stage)
  float* sZf = (float*)sH;
  const int tid = threadIdx.x;
  const int w = tid >> 5;
  const int row0 = blockIdx.x * 64;
  const int l = laneid();
  const int n = l & 15;
  const int mo = (l < 16) ? 0 : 8;

  if (tid < 32) tdm_load_64x64_f32(zup + (size_t)row0 * 64, sZf);
  __syncthreads();
  for (int i = tid; i < 64 * 64; i += 256) sZ[i] = (__bf16)sZf[i];
  __syncthreads();

  // GEMM1: z[64,64] @ dec_w1[64,256], relu
#pragma unroll
  for (int nn = 0; nn < 2; ++nn) {
    const int nt = w + nn * 8;
    const v16bf bf0 = frag_b_pk(pw1 + (size_t)(nt * 2 + 0) * 512);
    const v16bf bf1 = frag_b_pk(pw1 + (size_t)(nt * 2 + 1) * 512);
    const float bias = b1[nt * 16 + n];
#pragma unroll
    for (int mt = 0; mt < 4; ++mt) {
      v8f acc = {};
      acc = wmma_bf16(frag_a_bf16(sZ + mt * 16 * 64 + 0, 64), bf0, acc);
      acc = wmma_bf16(frag_a_bf16(sZ + mt * 16 * 64 + 32, 64), bf1, acc);
#pragma unroll
      for (int r = 0; r < 8; ++r) {
        float v = acc[r] + bias;
        v = v > 0.f ? v : 0.f;
        sH[(mt * 16 + mo + r) * HH + nt * 16 + n] = (__bf16)v;
      }
    }
  }
  __syncthreads();

  // GEMM2: H[64,256] @ dec_w2[256,64]; two m-tiles/wave share n-tile B frags
  {
    const int mt0 = w >> 2, mt1 = (w >> 2) + 2;
    const int nt = w & 3;
    const float bias = b2[nt * 16 + n];
    v8f acc0 = {}, acc1 = {};
#pragma unroll
    for (int kc = 0; kc < 8; ++kc) {
      const v16bf b = frag_b_pk(pw2 + (size_t)(nt * 8 + kc) * 512);
      acc0 = wmma_bf16(frag_a_bf16(sH + mt0 * 16 * HH + kc * 32, HH), b, acc0);
      acc1 = wmma_bf16(frag_a_bf16(sH + mt1 * 16 * HH + kc * 32, HH), b, acc1);
    }
#pragma unroll
    for (int r = 0; r < 8; ++r) {
      xrec[(size_t)(row0 + mt0 * 16 + mo + r) * 64 + nt * 16 + n] =
          1.f / (1.f + expf(-(acc0[r] + bias)));
      xrec[(size_t)(row0 + mt1 * 16 + mo + r) * 64 + nt * 16 + n] =
          1.f / (1.f + expf(-(acc1[r] + bias)));
    }
  }
}

// ===========================================================================
extern "C" void kernel_launch(void* const* d_in, const int* in_sizes, int n_in,
                              void* d_out, int out_size, void* d_ws, size_t ws_size,
                              hipStream_t stream) {
  (void)in_sizes; (void)n_in; (void)out_size; (void)ws_size;

  const float* x      = (const float*)d_in[0];
  const float* eps    = (const float*)d_in[1];
  const float* enc_w1 = (const float*)d_in[2];
  const float* enc_b1 = (const float*)d_in[3];
  const float* enc_w2 = (const float*)d_in[4];
  const float* enc_b2 = (const float*)d_in[5];
  const float* dec_w1 = (const float*)d_in[6];
  const float* dec_b1 = (const float*)d_in[7];
  const float* dec_w2 = (const float*)d_in[8];
  const float* dec_b2 = (const float*)d_in[9];
  const float* A      = (const float*)d_in[10];
  const float* Q      = (const float*)d_in[11];
  const float* Hm     = (const float*)d_in[12];
  const float* R      = (const float*)d_in[13];

  // Workspace (~33.8 MB): M/G/KT f32, packed bf16 weight fragments, U buffer.
  float* wsf  = (float*)d_ws;
  float* Mmat = wsf;                       // 4096 f
  float* Gmat = wsf + 4096;                // 4096 f
  float* KTm  = wsf + 8192;                // 4096 f
  __bf16* pk  = (__bf16*)(wsf + 12288);    // 90112 bf16 = 45056 f
  __bf16* pw1e = pk;                       // enc_w1: 16nt*2kc*512
  __bf16* pw2e = pk + 16384;               // enc_w2:  8nt*8kc*512
  __bf16* pw1d = pk + 49152;               // dec_w1: 16nt*2kc*512
  __bf16* pw2d = pk + 65536;               // dec_w2:  4nt*8kc*512
  __bf16* pG   = pk + 81920;               // G:       4nt*2kc*512
  __bf16* pKT  = pk + 86016;               // KT:      4nt*2kc*512
  float* U = wsf + 57344;                  // [NROWS,64] f32, reused as z_upd

  float* xrec  = (float*)d_out;
  float* zmean = xrec + (size_t)NROWS * 64;
  float* zlogv = zmean + (size_t)NROWS * 64;

  kvae_prep<<<1, 256, 0, stream>>>(A, Q, Hm, R, Mmat, Gmat, KTm);
  kvae_pack_b<<<32, 256, 0, stream>>>(enc_w1, pw1e, 16, 2, 256);
  kvae_pack_b<<<64, 256, 0, stream>>>(enc_w2, pw2e, 8, 8, 128);
  kvae_pack_b<<<32, 256, 0, stream>>>(dec_w1, pw1d, 16, 2, 256);
  kvae_pack_b<<<32, 256, 0, stream>>>(dec_w2, pw2d, 4, 8, 64);
  kvae_pack_b<<<8, 256, 0, stream>>>(Gmat, pG, 4, 2, 64);
  kvae_pack_b<<<8, 256, 0, stream>>>(KTm, pKT, 4, 2, 64);
  kvae_encoder<<<NROWS / 64, 256, 0, stream>>>(x, eps, enc_b1, enc_b2, pw1e, pw2e,
                                               pG, pKT, zmean, zlogv, U);
  kvae_scan<<<BB / 16, 128, 0, stream>>>(Mmat, U);
  kvae_decoder<<<NROWS / 64, 256, 0, stream>>>(U, dec_b1, dec_b2, pw1d, pw2d, xrec);
}
